// EDA_54211077210222
// MI455X (gfx1250) — matrix operand order repaired and checked by reading, not compiled, not run on hardware
//
#include <hip/hip_runtime.h>

typedef __attribute__((ext_vector_type(16))) _Float16 v16h;
typedef __attribute__((ext_vector_type(8)))  float    v8f;

#define HW_  16384   // 128*128
#define W_   128
#define B_   8
#define C2_  64

__device__ __forceinline__ int iabs_(int v) { return v < 0 ? -v : v; }

// --------------------------------------------------------------------------
// GEMM via WMMA: out[b, n, hw] = sum_k A[b, k, hw] * Wm[n, k] + bias[n]
// One wave owns one 16-pixel M-tile and ALL NT 16-wide N-tiles, so each
// staged A fragment feeds NT WMMAs (A traffic read exactly once).
// Weights are staged once per block into LDS as f16 (shared by all 8 waves).
// K, NT are compile-time so the K x NT WMMA loop fully unrolls.
// --------------------------------------------------------------------------
template<typename TIN, bool QUANT, int K, int NT>
__global__ __launch_bounds__(256)
void conv1x1_wmma(const TIN* __restrict__ A, const float* __restrict__ Wm,
                  const float* __restrict__ bias, float* __restrict__ out)
{
  constexpr int N = NT * 16;
  const int lane = threadIdx.x & 31;
  const int wave = threadIdx.x >> 5;

  __shared__ _Float16 sW[N * K];       // weights, f16, [n][k] row-major
  __shared__ _Float16 sA[8][32][16];   // per-wave A staging tile

  // cooperative weight stage (once per block)
  for (int i = threadIdx.x; i < N * K; i += 256)
    sW[i] = (_Float16)Wm[i];

  const long tile = (long)blockIdx.x * 8 + wave;   // M-tile index
  const long m0 = tile << 4;
  const int  b   = (int)(m0 / HW_);
  const int  hw0 = (int)(m0 % HW_);

  const int mloc = lane & 15;
  const int hi   = (lane >= 16) ? 1 : 0;

  const TIN* abase = A + ((long)b * K) * HW_ + hw0;

  v8f acc[NT];
  #pragma unroll
  for (int t = 0; t < NT; ++t) acc[t] = (v8f){};

  __syncthreads();   // sW ready

  for (int k0 = 0; k0 < K; k0 += 32) {
    // cooperative stage: lane l loads A row k0+l (16 contiguous pixels)
    const TIN* arow = abase + (long)(k0 + lane) * HW_;
    if (k0 + 32 < K)
      __builtin_prefetch(abase + (long)(k0 + 32 + lane) * HW_, 0, 0);
    _Float16* srow = &sA[wave][lane][0];
    #pragma unroll
    for (int j = 0; j < 16; ++j) srow[j] = (_Float16)(float)arow[j];
    __syncthreads();

    // A fragment: lanes 0-15 M=lane K{0..15},
    // lanes 16-31 M=lane-16 K{16..31}
    v16h af;
    #pragma unroll
    for (int i = 0; i < 16; ++i) {
      const int kl = hi * 16 + i;
      af[i] = sA[wave][kl][mloc];
    }

    // B fragments from LDS: lane n = n0 + (lane&15), halfs i -> K = k0 + 16*hi + i
    #pragma unroll
    for (int t = 0; t < NT; ++t) {
      const _Float16* wrow = &sW[(t * 16 + mloc) * K + hi * 16 + k0];
      v16h bf;
      #pragma unroll
      for (int i = 0; i < 16; ++i) bf[i] = wrow[i];
      acc[t] = __builtin_amdgcn_wmma_f32_16x16x32_f16(false, af, false, bf,
                                                      (short)0, acc[t], false, false);
    }
    __syncthreads();
  }

  // D layout: N = lane&15, VGPR r -> M = r + 8*hi  => contiguous 8 floats/lane
  #pragma unroll
  for (int t = 0; t < NT; ++t) {
    const int n = t * 16 + mloc;
    const float bv = bias[n];
    float* obase = out + ((long)b * N + n) * HW_ + hw0 + hi * 8;
    #pragma unroll
    for (int r = 0; r < 8; ++r) {
      float v = acc[t][r] + bv;
      if (QUANT) v = truncf(fminf(fmaxf(v, 0.0f), 255.0f)); // uint8 quantization
      obase[r] = v;
    }
  }
}

// --------------------------------------------------------------------------
// Canny per 128x128 map, entirely in LDS. One workgroup (256 threads) / map.
// feat layout [b, c, hw] (quantized floats), edge layout [(c*B+b), hw].
// --------------------------------------------------------------------------
__global__ __launch_bounds__(256)
void canny_kernel(const float* __restrict__ feat, float* __restrict__ edge)
{
  __shared__ unsigned char  su[HW_];    // quantized image (exact u8)
  __shared__ unsigned short smag[HW_];  // |gx|+|gy| <= 2040, exact
  __shared__ unsigned char  fl[HW_];    // bit0=weak, bit1=strongA, bit2=strongB

  const int map = blockIdx.x;
  const int c = map >> 3;       // / B_
  const int b = map & 7;        // % B_
  const float* in = feat + ((long)b * C2_ + c) * HW_;
  float* out = edge + (long)map * HW_;
  const int tid = threadIdx.x;

  for (int i = tid; i < HW_; i += 256) su[i] = (unsigned char)in[i];
  __syncthreads();

  // Sobel (replicate border), L1 magnitude
  for (int i = tid; i < HW_; i += 256) {
    const int y = i >> 7, x = i & 127;
    const int ym = (y > 0) ? y - 1 : 0, yp = (y < 127) ? y + 1 : 127;
    const int xm = (x > 0) ? x - 1 : 0, xp = (x < 127) ? x + 1 : 127;
    const int tl = su[ym * W_ + xm], tcv = su[ym * W_ + x], tr = su[ym * W_ + xp];
    const int ml = su[y * W_ + xm],                          mr = su[y * W_ + xp];
    const int bl = su[yp * W_ + xm], bcv = su[yp * W_ + x], br = su[yp * W_ + xp];
    const int gx = (tr + 2 * mr + br) - (tl + 2 * ml + bl);
    const int gy = (bl + 2 * bcv + br) - (tl + 2 * tcv + tr);
    smag[i] = (unsigned short)(iabs_(gx) + iabs_(gy));
  }
  __syncthreads();

  // NMS (zero-padded neighbors) + double threshold (low=80, high=300)
  for (int i = tid; i < HW_; i += 256) {
    const int y = i >> 7, x = i & 127;
    const int ym = (y > 0) ? y - 1 : 0, yp = (y < 127) ? y + 1 : 127;
    const int xm = (x > 0) ? x - 1 : 0, xp = (x < 127) ? x + 1 : 127;
    const int tl = su[ym * W_ + xm], tcv = su[ym * W_ + x], tr = su[ym * W_ + xp];
    const int ml = su[y * W_ + xm],                          mr = su[y * W_ + xp];
    const int bl = su[yp * W_ + xm], bcv = su[yp * W_ + x], br = su[yp * W_ + xp];
    const int gx = (tr + 2 * mr + br) - (tl + 2 * ml + bl);
    const int gy = (bl + 2 * bcv + br) - (tl + 2 * tcv + tr);
    const int mag = smag[i];
    const float ax = (float)iabs_(gx), ay = (float)iabs_(gy);
    const float TG22 = 0.4142135623730951f;
    const bool is_horiz = ay < TG22 * ax;
    const bool is_vert  = ax < TG22 * ay;
    const bool same_sign = (gx * gy) >= 0;

    auto nb = [&](int dy, int dx) -> int {
      const int yy = y + dy, xx = x + dx;
      return (yy >= 0 && yy < 128 && xx >= 0 && xx < 128)
                 ? (int)smag[yy * W_ + xx] : 0;
    };
    const bool keep_h  = (mag > nb(0, 1))  && (mag >= nb(0, -1));
    const bool keep_v  = (mag > nb(1, 0))  && (mag >= nb(-1, 0));
    const bool keep_d1 = (mag > nb(1, 1))  && (mag >= nb(-1, -1));
    const bool keep_d2 = (mag > nb(-1, 1)) && (mag >= nb(1, -1));
    const bool keep = is_horiz ? keep_h
                    : (is_vert ? keep_v : (same_sign ? keep_d1 : keep_d2));
    const bool strong = keep && (mag > 300);
    const bool weak   = keep && (mag > 80);
    fl[i] = (unsigned char)((weak ? 1 : 0) | (strong ? 2 : 0));
  }
  __syncthreads();

  // 16 synchronous hysteresis dilation iterations, double-buffered in bits 1/2
  const int dys[8] = {-1, -1, -1, 0, 0, 1, 1, 1};
  const int dxs[8] = {-1, 0, 1, -1, 1, -1, 0, 1};
  for (int it = 0; it < 16; ++it) {
    const int cb   = (it & 1) ? 4 : 2;
    const int nbit = (it & 1) ? 2 : 4;
    for (int i = tid; i < HW_; i += 256) {
      const int y = i >> 7, x = i & 127;
      const int f = fl[i];
      int nw = (f & cb) ? 1 : 0;
      if (!nw && (f & 1)) {
        int dil = 0;
        #pragma unroll
        for (int k = 0; k < 8; ++k) {
          const int yy = y + dys[k], xx = x + dxs[k];
          if (yy >= 0 && yy < 128 && xx >= 0 && xx < 128)
            dil |= fl[yy * W_ + xx] & cb;
        }
        nw = dil ? 1 : 0;
      }
      fl[i] = (unsigned char)((f & ~nbit) | (nw ? nbit : 0));
    }
    __syncthreads();
  }

  // after 16 iterations the result lives in bit 1 (S0)
  for (int i = tid; i < HW_; i += 256)
    out[i] = (fl[i] & 2) ? 255.0f : 0.0f;
}

// --------------------------------------------------------------------------
// aver[b, hw] = mean over c of edge[(c*B+b), hw]   (exact: integer sums)
// --------------------------------------------------------------------------
__global__ __launch_bounds__(256)
void aver_kernel(const float* __restrict__ edge, float* __restrict__ aver)
{
  const long gid = (long)blockIdx.x * 256 + threadIdx.x;  // over B*HW
  const int b = (int)(gid / HW_);
  const int hw = (int)(gid % HW_);
  float s = 0.0f;
  for (int c = 0; c < C2_; ++c) s += edge[((long)c * B_ + b) * HW_ + hw];
  aver[gid] = s * (1.0f / C2_);
}

// --------------------------------------------------------------------------
// Per-pixel: cov softmax over 64 channels for edge_c and edge_r, multiply
// with edge_n[b], emit f16 logits [b, c, hw].
// cov[b,c,h,w] = (e[c,B-1,w,h]-aver[b,w,h]) * (e[c,B-1,h,w]-aver[b,h,w]) / 64
// --------------------------------------------------------------------------
__global__ __launch_bounds__(256)
void softfuse_kernel(const float* __restrict__ ec, const float* __restrict__ er,
                     const float* __restrict__ en,
                     const float* __restrict__ avc, const float* __restrict__ avr,
                     _Float16* __restrict__ logits)
{
  const long gid = (long)blockIdx.x * 256 + threadIdx.x;  // over B*HW
  const int b = (int)(gid / HW_);
  const int hw = (int)(gid % HW_);
  const int h = hw >> 7, w = hw & 127;
  const int hwT = w * W_ + h;

  const float ac1 = avc[(long)b * HW_ + hw], ac2 = avc[(long)b * HW_ + hwT];
  const float ar1 = avr[(long)b * HW_ + hw], ar2 = avr[(long)b * HW_ + hwT];
  const long lastOff = (long)(B_ - 1) * HW_;

  auto covs = [&](int c, float& vc, float& vr) {
    const long base = (long)c * B_ * HW_ + lastOff;
    vc = (ec[base + hwT] - ac2) * (ec[base + hw] - ac1) * (1.0f / 64.0f);
    vr = (er[base + hwT] - ar2) * (er[base + hw] - ar1) * (1.0f / 64.0f);
  };

  float mc = -1e30f, mr = -1e30f;
  for (int c = 0; c < C2_; ++c) {
    float vc, vr; covs(c, vc, vr);
    mc = fmaxf(mc, vc); mr = fmaxf(mr, vr);
  }
  float sc = 0.0f, sr = 0.0f;
  for (int c = 0; c < C2_; ++c) {
    float vc, vr; covs(c, vc, vr);
    sc += __expf(vc - mc); sr += __expf(vr - mr);
  }
  const float isc = 1.0f / sc, isr = 1.0f / sr;
  for (int c = 0; c < C2_; ++c) {
    float vc, vr; covs(c, vc, vr);
    const float lg = en[((long)c * B_ + b) * HW_ + hw] *
                     (__expf(vc - mc) * isc) * (__expf(vr - mr) * isr);
    logits[((long)b * C2_ + c) * HW_ + hw] = (_Float16)lg;
  }
}

// --------------------------------------------------------------------------
extern "C" void kernel_launch(void* const* d_in, const int* in_sizes, int n_in,
                              void* d_out, int out_size, void* d_ws, size_t ws_size,
                              hipStream_t stream)
{
  (void)in_sizes; (void)n_in; (void)out_size; (void)ws_size;
  const float* x  = (const float*)d_in[0];
  const float* w1 = (const float*)d_in[1];
  const float* b1 = (const float*)d_in[2];
  const float* w2 = (const float*)d_in[3];
  const float* b2 = (const float*)d_in[4];
  const float* w3 = (const float*)d_in[5];
  const float* b3 = (const float*)d_in[6];
  const float* wo = (const float*)d_in[7];
  const float* bo = (const float*)d_in[8];

  char* ws = (char*)d_ws;
  const size_t F = (size_t)B_ * C2_ * HW_ * sizeof(float);   // 32 MB
  const size_t P = (size_t)B_ * HW_ * sizeof(float);         // 512 KB
  float*    feat   = (float*)(ws);
  float*    edge_c = (float*)(ws + F);
  float*    edge_n = (float*)(ws + 2 * F);
  float*    edge_r = (float*)(ws + 3 * F);
  float*    aver_c = (float*)(ws + 4 * F);
  float*    aver_r = (float*)(ws + 4 * F + P);
  _Float16* logits = (_Float16*)(ws + 4 * F + 2 * P);

  const dim3 blk(256);
  const int convBlocks = (B_ * HW_ / 16) / 8;   // 1024 (one M-tile per wave)
  const int mapBlocks = C2_ * B_;               // 512
  const int pixBlocks = B_ * HW_ / 256;         // 512

  conv1x1_wmma<float, true, 128, 4><<<convBlocks, blk, 0, stream>>>(x, w1, b1, feat);
  canny_kernel<<<mapBlocks, blk, 0, stream>>>(feat, edge_c);
  conv1x1_wmma<float, true, 128, 4><<<convBlocks, blk, 0, stream>>>(x, w2, b2, feat);
  canny_kernel<<<mapBlocks, blk, 0, stream>>>(feat, edge_n);
  conv1x1_wmma<float, true, 128, 4><<<convBlocks, blk, 0, stream>>>(x, w3, b3, feat);
  canny_kernel<<<mapBlocks, blk, 0, stream>>>(feat, edge_r);

  aver_kernel<<<pixBlocks, blk, 0, stream>>>(edge_c, aver_c);
  aver_kernel<<<pixBlocks, blk, 0, stream>>>(edge_r, aver_r);
  softfuse_kernel<<<pixBlocks, blk, 0, stream>>>(edge_c, edge_r, edge_n,
                                                 aver_c, aver_r, logits);
  conv1x1_wmma<_Float16, false, 64, 8><<<convBlocks, blk, 0, stream>>>(logits, wo, bo,
                                                                       (float*)d_out);
}
